// RWKV_Block_44220983280459
// MI455X (gfx1250) — matrix operand order repaired and checked by reading, not compiled, not run on hardware
//
#include <hip/hip_runtime.h>

// Problem constants (from reference): B=8, S=4096, H=1024
#define BB 8
#define SS 4096
#define HH 1024
#define MM (BB * SS)   // 32768 rows for all GEMMs

// Chunked-scan decomposition: S = CC chunks of LL
#define CC 64
#define LL 64          // CC * LL == SS

typedef __attribute__((ext_vector_type(16))) __bf16 bf16x16;
typedef __attribute__((ext_vector_type(8)))  __bf16 bf16x8;
typedef __attribute__((ext_vector_type(8)))  float  f32x8;

// ---------------------------------------------------------------------------
// Kernel 1: convert the three used weight matrices f32 -> bf16 (Wk is unused:
// the reference computes _k but never uses it in the output).
// ---------------------------------------------------------------------------
__global__ void cvt_weights(const float* __restrict__ wv,
                            const float* __restrict__ wr,
                            const float* __restrict__ wo,
                            __bf16* __restrict__ wv_b,
                            __bf16* __restrict__ wr_b,
                            __bf16* __restrict__ wo_b) {
    int i = blockIdx.x * blockDim.x + threadIdx.x;
    if (i < HH * HH) {
        wv_b[i] = (__bf16)wv[i];
        wr_b[i] = (__bf16)wr[i];
        wo_b[i] = (__bf16)wo[i];
    }
}

// ---------------------------------------------------------------------------
// Chunked linear scan: state_t = x_t + d * state_{t-1} is linear, so each
// chunk's contribution composes as  state_end = P_end + d^LL * carry_in.
//
// Pass 1: per (b,c,h) compute chunk-local partial end state E (zero init).
// 524288 threads = 16384 waves -> latency fully hidden.
// ---------------------------------------------------------------------------
__global__ void scan_pass1(const float* __restrict__ x,
                           const float* __restrict__ time_decay,
                           float* __restrict__ E) {
    int idx = blockIdx.x * blockDim.x + threadIdx.x;   // 0 .. B*C*H-1
    int h  = idx % HH;
    int bc = idx / HH;
    int c  = bc % CC;
    int b  = bc / CC;
    float d = __expf(-__expf(time_decay[h]));

    const float* xp = x + ((size_t)b * SS + (size_t)c * LL) * HH + h;
    float p = 0.0f;
#pragma unroll 4
    for (int i = 0; i < LL; ++i)
        p = xp[(size_t)i * HH] + p * d;
    E[((size_t)b * CC + c) * HH + h] = p;
}

// ---------------------------------------------------------------------------
// Pass 2: per (b,h) sequential scan over the 64 chunk partials (cheap, 2 MB):
//   carry_c (state entering chunk c) ;  carry_{c+1} = E_c + d^LL * carry_c
// d^LL = exp(-LL * exp(time_decay)) computed in closed form.
// ---------------------------------------------------------------------------
__global__ void scan_pass2(const float* __restrict__ time_decay,
                           const float* __restrict__ E,
                           float* __restrict__ Carr) {
    int idx = blockIdx.x * blockDim.x + threadIdx.x;   // 0 .. B*H-1
    int h = idx % HH;
    int b = idx / HH;
    float dL = __expf(-(float)LL * __expf(time_decay[h]));

    float carry = 0.0f;
    for (int c = 0; c < CC; ++c) {
        size_t off = ((size_t)b * CC + c) * HH + h;
        Carr[off] = carry;
        carry = E[off] + dL * carry;
    }
}

// ---------------------------------------------------------------------------
// Pass 3: per (b,c,h) re-walk the chunk from its known carry, fused with the
// time-mix and f32->bf16 conversion of both GEMM input matrices.
// ---------------------------------------------------------------------------
__global__ void scan_pass3(const float* __restrict__ x,
                           const float* __restrict__ time_decay,
                           const float* __restrict__ time_mix_v,
                           const float* __restrict__ time_mix_r,
                           const float* __restrict__ Carr,
                           __bf16* __restrict__ v_in,
                           __bf16* __restrict__ r_in) {
    int idx = blockIdx.x * blockDim.x + threadIdx.x;   // 0 .. B*C*H-1
    int h  = idx % HH;
    int bc = idx / HH;
    int c  = bc % CC;
    int b  = bc / CC;
    float d  = __expf(-__expf(time_decay[h]));
    float mv = time_mix_v[h];
    float mr = time_mix_r[h];
    float omv = 1.0f - mv, omr = 1.0f - mr;

    size_t base = ((size_t)b * SS + (size_t)c * LL) * HH + h;
    const float* xp = x    + base;
    __bf16*      vp = v_in + base;
    __bf16*      rp = r_in + base;

    float state = Carr[((size_t)b * CC + c) * HH + h];  // state_{s0-1}
#pragma unroll 4
    for (int i = 0; i < LL; ++i) {
        float xs   = xp[(size_t)i * HH];
        float prev = state;                              // state_{t-1} at step t
        vp[(size_t)i * HH] = (__bf16)(xs * mv + prev * omv);
        rp[(size_t)i * HH] = (__bf16)(xs * mr + prev * omr);
        state = xs + prev * d;
    }
}

// ---------------------------------------------------------------------------
// bf16 WMMA GEMM  C[M,N] = A[M,K] * W[N,K]^T   (M=32768, N=K=1024)
//
// Wave tile: 64x64 = 4x4 grid of v_wmma_f32_16x16x32_bf16.
// Block: 128 threads = 4 waves arranged 2x2 -> 128x128 block tile.
//
// Fragment addressing used here (wave32, contiguous k in both A and B):
//   A 16x32 bf16 : lane L -> row (L&15); VGPR0-3 = K[16*(L>>4) .. +7],
//                  VGPR4-7 = K[16*(L>>4) + 8 .. +7]  (two 16B loads)
//   B 32x16 bf16 : lane L -> col (L&15); VGPRs hold K[16*(L>>4) .. +15]
//                  (one contiguous 32B load — W rows are B columns)
//   C/D 16x16 f32: VGPR r -> row (r + 8*(L>>4)), col (L&15)
//
// MODE 0: store bf16 accumulator (v)
// MODE 1: store bf16 sigmoid(acc) (r)
// MODE 2: store f32  R[off] * acc  (final output)
// ---------------------------------------------------------------------------
template <int MODE>
__global__ __launch_bounds__(128)
void gemm_bf16_wmma(const __bf16* __restrict__ A,
                    const __bf16* __restrict__ W,
                    const __bf16* __restrict__ R,
                    void* __restrict__ Cout) {
    const int K = HH, N = HH;
    const int lane = threadIdx.x & 31;
    const int w    = threadIdx.x >> 5;
    const int lo   = lane & 15;
    const int hb   = lane >> 4;                 // 0 or 1 (half-wave)
    const int m0 = blockIdx.x * 128 + (w >> 1) * 64;
    const int n0 = blockIdx.y * 128 + (w & 1) * 64;

    f32x8 acc[4][4];
#pragma unroll
    for (int i = 0; i < 4; ++i)
#pragma unroll
        for (int j = 0; j < 4; ++j)
#pragma unroll
            for (int t = 0; t < 8; ++t)
                acc[i][j][t] = 0.0f;

    for (int kk = 0; kk < K; kk += 32) {
        bf16x16 af[4];
        bf16x16 bfm[4];
#pragma unroll
        for (int i = 0; i < 4; ++i) {
            const __bf16* ap = A + (size_t)(m0 + 16 * i + lo) * K + kk + hb * 16;
            bf16x8 a_lo = *(const bf16x8*)(ap);        // K = hb*16 .. +7
            bf16x8 a_hi = *(const bf16x8*)(ap + 8);    // K = hb*16 + 8 .. +7
            af[i] = __builtin_shufflevector(a_lo, a_hi,
                                            0, 1, 2, 3, 4, 5, 6, 7,
                                            8, 9, 10, 11, 12, 13, 14, 15);
        }
#pragma unroll
        for (int j = 0; j < 4; ++j) {
            const __bf16* bp = W + (size_t)(n0 + 16 * j + lo) * K + kk + hb * 16;
            bfm[j] = *(const bf16x16*)bp;              // 32B contiguous, 32B aligned
        }
#pragma unroll
        for (int i = 0; i < 4; ++i)
#pragma unroll
            for (int j = 0; j < 4; ++j)
                acc[i][j] = __builtin_amdgcn_wmma_f32_16x16x32_bf16(
                    false, af[i], false, bfm[j],
                    (short)0, acc[i][j], false, false);
    }

    // Epilogue
#pragma unroll
    for (int i = 0; i < 4; ++i) {
#pragma unroll
        for (int j = 0; j < 4; ++j) {
#pragma unroll
            for (int r = 0; r < 8; ++r) {
                int row = m0 + 16 * i + r + 8 * hb;
                int col = n0 + 16 * j + lo;
                size_t off = (size_t)row * N + col;
                float v = acc[i][j][r];
                if (MODE == 0) {
                    ((__bf16*)Cout)[off] = (__bf16)v;
                } else if (MODE == 1) {
                    ((__bf16*)Cout)[off] = (__bf16)(1.0f / (1.0f + __expf(-v)));
                } else {
                    ((float*)Cout)[off] = (float)R[off] * v;
                }
            }
        }
    }
}

// ---------------------------------------------------------------------------
// Launch: cvt weights -> 3-pass parallel scan/mix -> GEMM(v) -> GEMM(r,sig)
//         -> GEMM(out).
// Workspace layout (bytes):
//   [0,64M)     v_in  bf16 [32768,1024]
//   [64M,128M)  r_in  bf16
//   [128M,192M) v     bf16 (GEMM1 result, input to GEMM3)
//   [192M,256M) r     bf16 (sigmoid applied)
//   [256M,+6M)  Wv_b, Wr_b, Wo_b bf16
//   [+6M,+10M)  E, Carr f32 [B,CC,H] (2 MB each)
// ---------------------------------------------------------------------------
extern "C" void kernel_launch(void* const* d_in, const int* in_sizes, int n_in,
                              void* d_out, int out_size, void* d_ws, size_t ws_size,
                              hipStream_t stream) {
    (void)in_sizes; (void)n_in; (void)out_size; (void)ws_size;

    const float* x   = (const float*)d_in[0];
    const float* td  = (const float*)d_in[1];
    // d_in[2] = time_mix_k, d_in[5] = Wk : unused (k never reaches the output)
    const float* tmv = (const float*)d_in[3];
    const float* tmr = (const float*)d_in[4];
    const float* Wv  = (const float*)d_in[6];
    const float* Wr  = (const float*)d_in[7];
    const float* Wo  = (const float*)d_in[8];

    char* ws = (char*)d_ws;
    const size_t MH = (size_t)MM * HH;                 // 33,554,432 elements
    const size_t WB = (size_t)HH * HH;                 // 1,048,576 elements
    __bf16* v_in = (__bf16*)(ws);
    __bf16* r_in = (__bf16*)(ws + 2 * MH);
    __bf16* v_b  = (__bf16*)(ws + 4 * MH);
    __bf16* r_b  = (__bf16*)(ws + 6 * MH);
    __bf16* Wv_b = (__bf16*)(ws + 8 * MH);
    __bf16* Wr_b = (__bf16*)(ws + 8 * MH + 2 * WB);
    __bf16* Wo_b = (__bf16*)(ws + 8 * MH + 4 * WB);
    float*  E    = (float*)(ws + 8 * MH + 6 * WB);
    float*  Carr = (float*)(ws + 8 * MH + 6 * WB + 4 * (size_t)BB * CC * HH);

    cvt_weights<<<(HH * HH + 255) / 256, 256, 0, stream>>>(Wv, Wr, Wo, Wv_b, Wr_b, Wo_b);

    // 3-pass chunked scan (B*C*H = 524288 threads for passes 1 & 3)
    scan_pass1<<<(BB * CC * HH) / 256, 256, 0, stream>>>(x, td, E);
    scan_pass2<<<(BB * HH) / 256, 256, 0, stream>>>(td, E, Carr);
    scan_pass3<<<(BB * CC * HH) / 256, 256, 0, stream>>>(x, td, tmv, tmr, Carr, v_in, r_in);

    dim3 grid(MM / 128, HH / 128);   // 256 x 8 blocks, 128 threads (4 waves) each
    gemm_bf16_wmma<0><<<grid, 128, 0, stream>>>(v_in, Wv_b, nullptr, v_b);
    gemm_bf16_wmma<1><<<grid, 128, 0, stream>>>(r_in, Wr_b, nullptr, r_b);
    gemm_bf16_wmma<2><<<grid, 128, 0, stream>>>(v_b,  Wo_b, r_b,     d_out);
}
